// GPT2Attention_81922206204104
// MI455X (gfx1250) — matrix operand order repaired and checked by reading, not compiled, not run on hardware
//
#include <hip/hip_runtime.h>
#include <hip/hip_bf16.h>

typedef unsigned short u16;
typedef unsigned int   u32;
typedef unsigned long long u64;
typedef __attribute__((ext_vector_type(16))) __bf16 bf16x16;
typedef __attribute__((ext_vector_type(8)))  float  f32x8;
typedef __attribute__((ext_vector_type(4)))  u32    u32x4;
typedef __attribute__((ext_vector_type(8)))  int    i32x8;
typedef __attribute__((ext_vector_type(4)))  int    i32x4;

// ---- problem constants ----
#define Bb 2
#define Ss 2048
#define Ee 1024
#define Hh 16
#define Dd 64
#define Mtot (Bb * Ss)        // 4096
#define N_QKV (3 * Ee)        // 3072
#define HEADS_ELEMS ((size_t)Bb * Hh * Ss * Dd)

// ---------------- f32 -> bf16 conversion ----------------
static __device__ inline u16 f2bf_sw(float f) {
    u32 u = __float_as_uint(f);
    u += 0x7FFFu + ((u >> 16) & 1u);
    return (u16)(u >> 16);
}
// packed pair of bf16 in one u32 (low = a, high = b)
static __device__ inline u32 pack2(float a, float b) {
#if __has_builtin(__builtin_amdgcn_cvt_pk_bf16_f32)
    auto p = __builtin_amdgcn_cvt_pk_bf16_f32(a, b);
    union { decltype(p) v; u32 u; } cv; cv.v = p;
    return cv.u;
#elif __has_builtin(__builtin_amdgcn_perm)
    // round-half-up + byte-perm: 3 VALU ops per pair
    const u32 ua = __float_as_uint(a) + 0x8000u;
    const u32 ub = __float_as_uint(b) + 0x8000u;
    return __builtin_amdgcn_perm(ub, ua, 0x07060302u);  // {ub[3],ub[2],ua[3],ua[2]}
#else
    return (u32)f2bf_sw(a) | ((u32)f2bf_sw(b) << 16);
#endif
}
static __device__ inline u16 bf_lo(u32 p) { return (u16)(p & 0xFFFFu); }
static __device__ inline u16 bf_hi(u32 p) { return (u16)(p >> 16); }

union FragU { bf16x16 v; uint4 u[2]; };
static __device__ inline bf16x16 ld_frag(const u16* p0, const u16* p1) {
    FragU f;
    f.u[0] = *reinterpret_cast<const uint4*>(p0);
    f.u[1] = *reinterpret_cast<const uint4*>(p1);
    return f.v;
}
static __device__ inline f32x8 wmma_bf16(bf16x16 a, bf16x16 b, f32x8 c) {
    return __builtin_amdgcn_wmma_f32_16x16x32_bf16(false, a, false, b, (short)0, c, false, false);
}

// ---------------- Tensor Data Mover helpers ----------------
// 2D tile load: rows x 64 bf16 elements, row stride 64, packed into LDS.
static __device__ inline void tdm_load_k_tile(u32 lds_addr, const u16* gptr,
                                              u32 rem_rows, u32 tile_rows) {
    const u64 ga = (u64)(uintptr_t)gptr;
    u32x4 g0; i32x8 g1; i32x4 g2 = {0, 0, 0, 0}, g3 = {0, 0, 0, 0};
    g0[0] = 1u;                                    // count=1, user mode
    g0[1] = lds_addr;                              // lds_addr
    g0[2] = (u32)ga;                               // global_addr[31:0]
    g0[3] = (u32)((ga >> 32) & 0x01FFFFFFu) | (2u << 30);   // addr[56:32] | type=2
    g1[0] = (int)(1u << 16);                       // wg_mask=0, data_size=2B
    g1[1] = (int)(64u << 16);                      // tensor_dim0 = 64
    g1[2] = (int)((rem_rows & 0xFFFFu) << 16);     // tensor_dim1 lo16
    g1[3] = (int)((rem_rows >> 16) | (64u << 16)); // tensor_dim1 hi16 | tile_dim0=64
    g1[4] = (int)tile_rows;                        // tile_dim1, tile_dim2=0
    g1[5] = (int)64u;                              // tensor_dim0_stride = 64
    g1[6] = 0;
    g1[7] = 0;
#if __has_builtin(__builtin_amdgcn_tensor_load_to_lds)
#if defined(__clang_major__) && (__clang_major__ <= 22)
    __builtin_amdgcn_tensor_load_to_lds(g0, g1, g2, g3, 0);
#else
    i32x8 g4 = {0, 0, 0, 0, 0, 0, 0, 0};
    __builtin_amdgcn_tensor_load_to_lds(g0, g1, g2, g3, g4, 0);
#endif
#else
    asm volatile("tensor_load_to_lds %0, %1, %2, %3"
                 :: "s"(g0), "s"(g1), "s"(g2), "s"(g3) : "memory");
#endif
}
static __device__ inline void wait_tensorcnt0() {
#if __has_builtin(__builtin_amdgcn_s_wait_tensorcnt)
    __builtin_amdgcn_s_wait_tensorcnt(0);
#else
    asm volatile("s_wait_tensorcnt 0x0" ::: "memory");
#endif
}

// =====================================================================
// Kernel 1: QKV GEMM.  X[4096,1024]f32 @ W[1024,3072]f32 + b -> bf16
// scattered to Q/K/V [B,H,S,D].  Block tile 128x64, wave tile 32x32.
// =====================================================================
__global__ __launch_bounds__(256)
void qkv_gemm_kernel(const float* __restrict__ X, const float* __restrict__ W,
                     const float* __restrict__ bias, u16* __restrict__ qkv_ws) {
    __shared__ u16 As[128][32];
    __shared__ u16 Bs[64][32];

    const int tid  = threadIdx.x;
    const int lane = tid & 31;
    const int wave = tid >> 5;
    const int wm   = wave & 3;
    const int wn   = wave >> 2;
    const int half = lane >> 4;
    const int l15  = lane & 15;

    const int m0 = blockIdx.y * 128;
    const int n0 = blockIdx.x * 64;

    f32x8 c00 = {}, c01 = {}, c10 = {}, c11 = {};

    const int arow0 = wm * 32 + l15;
    const int arow1 = arow0 + 16;
    const int ac0   = half ? 8 : 0;
    const int bk0   = half ? 8 : 0;
    const int bcol0 = wn * 32 + l15;
    const int bcol1 = bcol0 + 16;

    for (int k0 = 0; k0 < Ee; k0 += 32) {
        {   // ---- stage A: 128x32 f32 -> bf16, 16 elems/thread ----
            const int e  = tid * 16;
            const int r  = e >> 5;
            const int cc = e & 31;
            const float* src = X + (size_t)(m0 + r) * Ee + k0 + cc;
            float4 x0 = *reinterpret_cast<const float4*>(src);
            float4 x1 = *reinterpret_cast<const float4*>(src + 4);
            float4 x2 = *reinterpret_cast<const float4*>(src + 8);
            float4 x3 = *reinterpret_cast<const float4*>(src + 12);
            uint4 w0, w1;
            w0.x = pack2(x0.x, x0.y); w0.y = pack2(x0.z, x0.w);
            w0.z = pack2(x1.x, x1.y); w0.w = pack2(x1.z, x1.w);
            w1.x = pack2(x2.x, x2.y); w1.y = pack2(x2.z, x2.w);
            w1.z = pack2(x3.x, x3.y); w1.w = pack2(x3.z, x3.w);
            *reinterpret_cast<uint4*>(&As[r][cc])     = w0;
            *reinterpret_cast<uint4*>(&As[r][cc + 8]) = w1;
            if (k0 + 32 < Ee) __builtin_prefetch(src + 32, 0, 0);
        }
        {   // ---- stage B transposed ----
            const int e = tid * 8;
            const int k = e >> 6;
            const int n = e & 63;
            const float* src = W + (size_t)(k0 + k) * N_QKV + n0 + n;
            float4 x0 = *reinterpret_cast<const float4*>(src);
            float4 x1 = *reinterpret_cast<const float4*>(src + 4);
            u32 p0 = pack2(x0.x, x0.y), p1 = pack2(x0.z, x0.w);
            u32 p2 = pack2(x1.x, x1.y), p3 = pack2(x1.z, x1.w);
            Bs[n + 0][k] = bf_lo(p0); Bs[n + 1][k] = bf_hi(p0);
            Bs[n + 2][k] = bf_lo(p1); Bs[n + 3][k] = bf_hi(p1);
            Bs[n + 4][k] = bf_lo(p2); Bs[n + 5][k] = bf_hi(p2);
            Bs[n + 6][k] = bf_lo(p3); Bs[n + 7][k] = bf_hi(p3);
        }
        __syncthreads();

        bf16x16 fa0 = ld_frag(&As[arow0][ac0], &As[arow0][ac0 + 16]);
        bf16x16 fa1 = ld_frag(&As[arow1][ac0], &As[arow1][ac0 + 16]);
        bf16x16 fb0 = ld_frag(&Bs[bcol0][bk0], &Bs[bcol0][bk0 + 16]);
        bf16x16 fb1 = ld_frag(&Bs[bcol1][bk0], &Bs[bcol1][bk0 + 16]);
        c00 = wmma_bf16(fa0, fb0, c00);
        c01 = wmma_bf16(fa0, fb1, c01);
        c10 = wmma_bf16(fa1, fb0, c10);
        c11 = wmma_bf16(fa1, fb1, c11);
        __syncthreads();
    }

#pragma unroll
    for (int ti = 0; ti < 2; ++ti) {
#pragma unroll
        for (int r = 0; r < 8; ++r) {
            const int m  = m0 + wm * 32 + ti * 16 + r + half * 8;
            const int nA = n0 + wn * 32 + l15;
            const int nB = nA + 16;
            const float vA = (ti ? c10[r] : c00[r]) + bias[nA];
            const float vB = (ti ? c11[r] : c01[r]) + bias[nB];
            const u32 p = pack2(vA, vB);
            const int b = m >> 11, s = m & 2047;
#pragma unroll
            for (int hn = 0; hn < 2; ++hn) {
                const int n = hn ? nB : nA;
                const int which = n >> 10;
                const int e = n & 1023;
                const int h = e >> 6, d = e & 63;
                qkv_ws[(size_t)which * HEADS_ELEMS +
                       ((((size_t)b * Hh + h) * Ss + s) * Dd + d)] = hn ? bf_hi(p) : bf_lo(p);
            }
        }
    }
}

// =====================================================================
// Kernel 2: causal flash attention per (b,h).  K tile streamed via TDM.
// =====================================================================
__global__ __launch_bounds__(256)
void attn_kernel(const u16* __restrict__ Q, const u16* __restrict__ K,
                 const u16* __restrict__ V, u16* __restrict__ AO) {
    __shared__ u16 Ks[32][64];      // [key][d]   (TDM destination)
    __shared__ u16 Vt[64][32];      // [d][key]
    __shared__ u16 Ps[8][16][32];   // per-wave P scratch

    const int tid  = threadIdx.x;
    const int lane = tid & 31;
    const int wave = tid >> 5;
    const int half = lane >> 4;
    const int l15  = lane & 15;

    const int bh = blockIdx.y;
    const u16* Qp = Q + (size_t)bh * Ss * Dd;
    const u16* Kp = K + (size_t)bh * Ss * Dd;
    const u16* Vp = V + (size_t)bh * Ss * Dd;

    const int qrow0 = blockIdx.x * 128;
    const int qbase = qrow0 + wave * 16;

    const int qr  = qbase + l15;
    const int ac0 = half ? 8 : 0;
    bf16x16 fq0 = ld_frag(Qp + (size_t)qr * Dd + ac0,      Qp + (size_t)qr * Dd + ac0 + 16);
    bf16x16 fq1 = ld_frag(Qp + (size_t)qr * Dd + 32 + ac0, Qp + (size_t)qr * Dd + 32 + ac0 + 16);

    f32x8 o[4] = {{}, {}, {}, {}};
    float m_r[8], l_r[8];
#pragma unroll
    for (int r = 0; r < 8; ++r) { m_r[r] = -1e30f; l_r[r] = 0.0f; }

    const u32 kls = (u32)(uintptr_t)(&Ks[0][0]);

    const int nkb = blockIdx.x * 4 + 4;
    for (int jb = 0; jb < nkb; ++jb) {
        const int j0 = jb * 32;

        // ---- K tile: Tensor Data Mover DMA (wave 0 issues) ----
        if (wave == 0) {
            tdm_load_k_tile(kls, Kp + (size_t)j0 * Dd, (u32)(Ss - j0), 32u);
        }
        // ---- V tile transposed (all threads) ----
        {
            const int e   = tid * 8;
            const int kr  = e >> 6;
            const int dcc = e & 63;
            union { uint4 u; u16 s[8]; } vt;
            vt.u = *reinterpret_cast<const uint4*>(Vp + (size_t)(j0 + kr) * Dd + dcc);
#pragma unroll
            for (int i = 0; i < 8; ++i) Vt[dcc + i][kr] = vt.s[i];
        }
        if (wave == 0) wait_tensorcnt0();
        __syncthreads();

        // ---- scores: Q @ K^T ----
        f32x8 s0 = {}, s1 = {};
        {
            const int bk0 = half ? 8 : 0;
            const int kc0 = l15;
            const int kc1 = 16 + l15;
            bf16x16 fk;
            fk = ld_frag(&Ks[kc0][bk0],      &Ks[kc0][bk0 + 16]);      s0 = wmma_bf16(fq0, fk, s0);
            fk = ld_frag(&Ks[kc0][32 + bk0], &Ks[kc0][32 + bk0 + 16]); s0 = wmma_bf16(fq1, fk, s0);
            fk = ld_frag(&Ks[kc1][bk0],      &Ks[kc1][bk0 + 16]);      s1 = wmma_bf16(fq0, fk, s1);
            fk = ld_frag(&Ks[kc1][32 + bk0], &Ks[kc1][32 + bk0 + 16]); s1 = wmma_bf16(fq1, fk, s1);
        }

        // ---- scale + causal mask ----
#pragma unroll
        for (int r = 0; r < 8; ++r) {
            const int row = qbase + r + half * 8;
            float v0 = s0[r] * 0.125f;
            float v1 = s1[r] * 0.125f;
            if (j0 + l15 > row)      v0 = -1e30f;
            if (j0 + 16 + l15 > row) v1 = -1e30f;
            s0[r] = v0; s1[r] = v1;
        }

        // ---- online softmax ----
#pragma unroll
        for (int r = 0; r < 8; ++r) {
            float mx = fmaxf(s0[r], s1[r]);
#pragma unroll
            for (int msk = 1; msk < 16; msk <<= 1)
                mx = fmaxf(mx, __shfl_xor(mx, msk, 32));
            const float mnew  = fmaxf(m_r[r], mx);
            const float scale = __expf(m_r[r] - mnew);
            const float p0 = __expf(s0[r] - mnew);
            const float p1 = __expf(s1[r] - mnew);
            float rs = p0 + p1;
#pragma unroll
            for (int msk = 1; msk < 16; msk <<= 1)
                rs += __shfl_xor(rs, msk, 32);
            l_r[r] = l_r[r] * scale + rs;
            m_r[r] = mnew;
#pragma unroll
            for (int t = 0; t < 4; ++t) o[t][r] *= scale;
            const u32 pp = pack2(p0, p1);
            const int prow = r + half * 8;
            Ps[wave][prow][l15]      = bf_lo(pp);
            Ps[wave][prow][16 + l15] = bf_hi(pp);
        }
        asm volatile("s_wait_dscnt 0" ::: "memory");

        // ---- P @ V ----
        {
            const int pc0 = half ? 8 : 0;
            bf16x16 fp = ld_frag(&Ps[wave][l15][pc0], &Ps[wave][l15][pc0 + 16]);
            const int vk0 = half ? 8 : 0;
#pragma unroll
            for (int t = 0; t < 4; ++t) {
                const int dcol = t * 16 + l15;
                bf16x16 fv = ld_frag(&Vt[dcol][vk0], &Vt[dcol][vk0 + 16]);
                o[t] = wmma_bf16(fp, fv, o[t]);
            }
        }
        __syncthreads();
    }

    // ---- epilogue ----
    const int h = bh & 15;
    const int b = bh >> 4;
#pragma unroll
    for (int r = 0; r < 8; ++r) {
        const float rinv = 1.0f / l_r[r];
        const int row = qbase + r + half * 8;
#pragma unroll
        for (int t = 0; t < 4; t += 2) {
            const u32 p = pack2(o[t][r] * rinv, o[t + 1][r] * rinv);
            const int col = h * 64 + t * 16 + l15;
            AO[((size_t)b * Ss + row) * Ee + col]      = bf_lo(p);
            AO[((size_t)b * Ss + row) * Ee + col + 16] = bf_hi(p);
        }
    }
}

// =====================================================================
// Kernel 3: output projection.  AO bf16 @ Wp f32 + b -> f32.
// =====================================================================
__global__ __launch_bounds__(256)
void proj_gemm_kernel(const u16* __restrict__ A, const float* __restrict__ W,
                      const float* __restrict__ bias, float* __restrict__ out) {
    __shared__ u16 As[128][32];
    __shared__ u16 Bs[64][32];

    const int tid  = threadIdx.x;
    const int lane = tid & 31;
    const int wave = tid >> 5;
    const int wm   = wave & 3;
    const int wn   = wave >> 2;
    const int half = lane >> 4;
    const int l15  = lane & 15;

    const int m0 = blockIdx.y * 128;
    const int n0 = blockIdx.x * 64;

    f32x8 c00 = {}, c01 = {}, c10 = {}, c11 = {};

    const int arow0 = wm * 32 + l15;
    const int arow1 = arow0 + 16;
    const int ac0   = half ? 8 : 0;
    const int bk0   = half ? 8 : 0;
    const int bcol0 = wn * 32 + l15;
    const int bcol1 = bcol0 + 16;

    for (int k0 = 0; k0 < Ee; k0 += 32) {
        {   // A already bf16
            const int e  = tid * 16;
            const int r  = e >> 5;
            const int cc = e & 31;
            const u16* src = A + (size_t)(m0 + r) * Ee + k0 + cc;
            *reinterpret_cast<uint4*>(&As[r][cc])     = *reinterpret_cast<const uint4*>(src);
            *reinterpret_cast<uint4*>(&As[r][cc + 8]) = *reinterpret_cast<const uint4*>(src + 8);
            if (k0 + 32 < Ee) __builtin_prefetch(src + 32, 0, 0);
        }
        {   // B: f32 -> bf16 transposed
            const int e = tid * 8;
            const int k = e >> 6;
            const int n = e & 63;
            const float* src = W + (size_t)(k0 + k) * Ee + n0 + n;
            float4 x0 = *reinterpret_cast<const float4*>(src);
            float4 x1 = *reinterpret_cast<const float4*>(src + 4);
            u32 p0 = pack2(x0.x, x0.y), p1 = pack2(x0.z, x0.w);
            u32 p2 = pack2(x1.x, x1.y), p3 = pack2(x1.z, x1.w);
            Bs[n + 0][k] = bf_lo(p0); Bs[n + 1][k] = bf_hi(p0);
            Bs[n + 2][k] = bf_lo(p1); Bs[n + 3][k] = bf_hi(p1);
            Bs[n + 4][k] = bf_lo(p2); Bs[n + 5][k] = bf_hi(p2);
            Bs[n + 6][k] = bf_lo(p3); Bs[n + 7][k] = bf_hi(p3);
        }
        __syncthreads();

        bf16x16 fa0 = ld_frag(&As[arow0][ac0], &As[arow0][ac0 + 16]);
        bf16x16 fa1 = ld_frag(&As[arow1][ac0], &As[arow1][ac0 + 16]);
        bf16x16 fb0 = ld_frag(&Bs[bcol0][bk0], &Bs[bcol0][bk0 + 16]);
        bf16x16 fb1 = ld_frag(&Bs[bcol1][bk0], &Bs[bcol1][bk0 + 16]);
        c00 = wmma_bf16(fa0, fb0, c00);
        c01 = wmma_bf16(fa0, fb1, c01);
        c10 = wmma_bf16(fa1, fb0, c10);
        c11 = wmma_bf16(fa1, fb1, c11);
        __syncthreads();
    }

#pragma unroll
    for (int ti = 0; ti < 2; ++ti) {
#pragma unroll
        for (int r = 0; r < 8; ++r) {
            const int m = m0 + wm * 32 + ti * 16 + r + half * 8;
            const int n = n0 + wn * 32 + l15;
            out[(size_t)m * Ee + n]      = (ti ? c10[r] : c00[r]) + bias[n];
            out[(size_t)m * Ee + n + 16] = (ti ? c11[r] : c01[r]) + bias[n + 16];
        }
    }
}

// =====================================================================
extern "C" void kernel_launch(void* const* d_in, const int* in_sizes, int n_in,
                              void* d_out, int out_size, void* d_ws, size_t ws_size,
                              hipStream_t stream) {
    const float* hs     = (const float*)d_in[0];
    const float* attn_w = (const float*)d_in[1];
    const float* attn_b = (const float*)d_in[2];
    const float* proj_w = (const float*)d_in[3];
    const float* proj_b = (const float*)d_in[4];
    float* out = (float*)d_out;

    u16* wsp = (u16*)d_ws;
    u16* Qw  = wsp;
    u16* Kw  = wsp + HEADS_ELEMS;
    u16* Vw  = wsp + 2 * HEADS_ELEMS;
    u16* AOw = wsp + 3 * HEADS_ELEMS;
    (void)in_sizes; (void)n_in; (void)out_size; (void)ws_size;

    qkv_gemm_kernel<<<dim3(N_QKV / 64, Mtot / 128), 256, 0, stream>>>(hs, attn_w, attn_b, wsp);
    attn_kernel<<<dim3(Ss / 128, Bb * Hh), 256, 0, stream>>>(Qw, Kw, Vw, AOw);
    proj_gemm_kernel<<<dim3(Ee / 64, Mtot / 128), 256, 0, stream>>>(AOw, proj_w, proj_b, out);
}
